// RefInvTorchNaive_52639119179855
// MI455X (gfx1250) — matrix operand order repaired and checked by reading, not compiled, not run on hardware
//
#include <hip/hip_runtime.h>
#include <stdint.h>

namespace {
constexpr int   kB = 4, kN = 4096, kE = 128;
constexpr int   kNK = 4, kCache = 32, kCE = 32;
constexpr int   kM = kN / kCache;          // 128 groups
constexpr float kAlpha = 0.5f, kMom = 0.9f;
constexpr int   kLdsStride = 36;           // dwords/row: 32 data + 4 pad, keeps 16B alignment
}

typedef __attribute__((ext_vector_type(16))) __bf16 v16bf;
typedef __attribute__((ext_vector_type(8)))  float  v8f;
typedef __attribute__((ext_vector_type(8)))  unsigned v8u;

// gfx1250 hardware tanh (TRANS unit, co-executes with VALU)
__device__ __forceinline__ float hw_tanh(float x) {
#if __has_builtin(__builtin_amdgcn_tanhf)
  return __builtin_amdgcn_tanhf(x);
#else
  float r;
  asm("v_tanh_f32 %0, %1" : "=v"(r) : "v"(x));
  return r;
#endif
}

// broadcast lane j's value to all lanes via v_readlane (j is wave-uniform)
__device__ __forceinline__ float bcast_lane(float v, int j) {
  return __int_as_float(__builtin_amdgcn_readlane(__float_as_int(v), j));
}

// pack two f32 into a dword of two bf16 (RNE)
__device__ __forceinline__ unsigned pack_bf16(float a, float b) {
  union { __bf16 h[2]; unsigned u; } c;
  c.h[0] = (__bf16)a; c.h[1] = (__bf16)b;
  return c.u;
}

// CDNA5 async global->LDS copy (ASYNCcnt-tracked, no VGPR round trip)
__device__ __forceinline__ void async_tile_load_b128(const float* gsrc, unsigned lds_byte_off) {
  asm volatile("global_load_async_to_lds_b128 %0, %1, off"
               :: "v"(lds_byte_off), "v"(gsrc)
               : "memory");
}

__global__ __launch_bounds__(32) void ncn_layer_kernel(
    const float* __restrict__ x_in, const float* __restrict__ xa_in,
    const float* __restrict__ Wl,                 // this layer's [2E] weights
    float* __restrict__ x_out, float* __restrict__ xa_out,
    int sstep) {                                   // group stride: 0 (stage0) / 1 (stage1)
  __shared__ __align__(16) float xs[kCache * kLdsStride];
  __shared__ __align__(16) float as_[kCache * kLdsStride];

  const int  lane   = threadIdx.x;
  const bool hiHalf = lane >= 16;
  const int  n16    = lane & 15;                   // column / row-pair owner
  const int  eB     = hiHalf ? 16 : 0;             // this lane's e-range base
  int bid = blockIdx.x;
  const int n = bid % kNK;  bid /= kNK;            // kernel-feature slice (uniform per wave)
  const int g = bid % kM;   bid /= kM;             // group
  const int b = bid;                               // batch

  // ---- stage both [32 x 32] f32 tiles into LDS, coalesced (8 lanes == one 128B row) ----
  const unsigned xs_base = (unsigned)(size_t)&xs[0];   // low 32 bits of flat LDS addr = offset
  const unsigned as_base = (unsigned)(size_t)&as_[0];
  #pragma unroll
  for (int i = 0; i < 8; ++i) {
    const int r   = i * 4 + (lane >> 3);                           // tile row 0..31
    const int row = ((g + r * sstep) % kM) * kCache + r;           // permuted global row
    const size_t goff  = ((size_t)b * kN + row) * kE + n * kCE + (size_t)(lane & 7) * 4;
    const unsigned off = (unsigned)(r * kLdsStride + (lane & 7) * 4) * 4u;
    async_tile_load_b128(x_in  + goff, xs_base + off);
    async_tile_load_b128(xa_in + goff, as_base + off);
  }
  asm volatile("s_wait_asynccnt 0" ::: "memory");

  // ---- half-row register layout: lane holds e[eB..eB+16) of rows n16 and 16+n16 ----
  // This makes the bf16 WMMA B-operand layout native (no per-step cross-lane ops).
  float xiA[16], xiB[16], xaA[16], xaB[16];
  #pragma unroll
  for (int c = 0; c < 4; ++c) {
    const float4 t0 = *reinterpret_cast<const float4*>(&xs[n16 * kLdsStride + eB + c * 4]);
    xiA[4*c+0] = t0.x; xiA[4*c+1] = t0.y; xiA[4*c+2] = t0.z; xiA[4*c+3] = t0.w;
    const float4 t1 = *reinterpret_cast<const float4*>(&xs[(16 + n16) * kLdsStride + eB + c * 4]);
    xiB[4*c+0] = t1.x; xiB[4*c+1] = t1.y; xiB[4*c+2] = t1.z; xiB[4*c+3] = t1.w;
    const float4 u0 = *reinterpret_cast<const float4*>(&as_[n16 * kLdsStride + eB + c * 4]);
    xaA[4*c+0] = u0.x; xaA[4*c+1] = u0.y; xaA[4*c+2] = u0.z; xaA[4*c+3] = u0.w;
    const float4 u1 = *reinterpret_cast<const float4*>(&as_[(16 + n16) * kLdsStride + eB + c * 4]);
    xaB[4*c+0] = u1.x; xaB[4*c+1] = u1.y; xaB[4*c+2] = u1.z; xaB[4*c+3] = u1.w;
  }

  // ---- weights for this kernel slice: n is uniform -> scalar loads ----
  float wif[kCE], wjf[kCE];
  #pragma unroll
  for (int e = 0; e < kCE; ++e) {
    wif[e] = Wl[n * kCE + e];
    wjf[e] = Wl[kE + n * kCE + e];
  }

  // ---- A operands (built once): weight vector broadcast into ALL 16 rows, so the
  // C output is row-constant: C[0] at lane l == dot for column (l & 15) in BOTH halves.
  // A k order used here (contiguous, same as B): vgpr v: lanes0-15 K-pair klo=2v; lanes16-31 klo+16.
  v8u a1_dw, a2_dw;
  #pragma unroll
  for (int v = 0; v < 8; ++v) {
    const int klo = 2 * v;
    const unsigned wi_l = pack_bf16(wif[klo],     wif[klo + 1]);
    const unsigned wi_h = pack_bf16(wif[klo + 16], wif[klo + 17]);
    const unsigned wj_l = pack_bf16(wjf[klo],     wjf[klo + 1]);
    const unsigned wj_h = pack_bf16(wjf[klo + 16], wjf[klo + 17]);
    a1_dw[v] = hiHalf ? wi_h : wi_l;
    a2_dw[v] = hiHalf ? wj_h : wj_l;
  }
  const v16bf A1 = __builtin_bit_cast(v16bf, a1_dw);   // Wi rows
  const v16bf A2 = __builtin_bit_cast(v16bf, a2_dw);   // Wj rows

  // ---- 32-step serial in-register scan ----
  #pragma unroll 1
  for (int j = 0; j < kCache; ++j) {
    // B operands straight from the half-row slots (native layout, pure per-lane packs):
    // B_lo columns = rows 0..15 (slot A), B_hi columns = rows 16..31 (slot B).
    v8u blo_dw, bhi_dw;
    #pragma unroll
    for (int v = 0; v < 8; ++v) {
      blo_dw[v] = pack_bf16(xiA[2 * v], xiA[2 * v + 1]);
      bhi_dw[v] = pack_bf16(xiB[2 * v], xiB[2 * v + 1]);
    }
    const v16bf Blo = __builtin_bit_cast(v16bf, blo_dw);
    const v16bf Bhi = __builtin_bit_cast(v16bf, bhi_dw);
    const v8f cz = {};
    const v8f c1lo = __builtin_amdgcn_wmma_f32_16x16x32_bf16(false, A1, false, Blo, (short)0, cz, false, false);
    const v8f c1hi = __builtin_amdgcn_wmma_f32_16x16x32_bf16(false, A1, false, Bhi, (short)0, cz, false, false);
    const v8f c2lo = __builtin_amdgcn_wmma_f32_16x16x32_bf16(false, A2, false, Blo, (short)0, cz, false, false);
    const v8f c2hi = __builtin_amdgcn_wmma_f32_16x16x32_bf16(false, A2, false, Bhi, (short)0, cz, false, false);

    const float d1A = c1lo[0];                 // d1[n16]      (both halves, no shuffle)
    const float d1B = c1hi[0];                 // d1[16+n16]
    const float s2  = bcast_lane((j < 16) ? c2lo[0] : c2hi[0], j & 15);   // d2[row j]

    // snapshot of row j (pre-update, f32): row j lives in slot A (j<16) or B (j>=16),
    // split across lanes (j&15) [e 0..15] and (j&15)+16 [e 16..31].
    float xjsel[16];
    #pragma unroll
    for (int t = 0; t < 16; ++t) {
      const float src = (j < 16) ? xiA[t] : xiB[t];
      const float lo  = bcast_lane(src, j & 15);          // xj[t]
      const float hi  = bcast_lane(src, (j & 15) + 16);   // xj[16+t]
      xjsel[t] = hiHalf ? hi : lo;                        // xj[eB + t] for this lane
    }

    const float hA = (1.0f - kAlpha) * (d1A + s2);
    const float hB = (1.0f - kAlpha) * (d1B + s2);
    #pragma unroll
    for (int t = 0; t < 16; ++t) {
      float T = fmaf(hA, xjsel[t], kAlpha * xiA[t]);      // row n16
      float F = hw_tanh(T);
      xaA[t] = fmaf(kMom, xaA[t], (1.0f - kMom) * F);
      xiA[t] = fmaf(kMom, xiA[t], (1.0f - kMom) * xaA[t]);
      T = fmaf(hB, xjsel[t], kAlpha * xiB[t]);            // row 16+n16
      F = hw_tanh(T);
      xaB[t] = fmaf(kMom, xaB[t], (1.0f - kMom) * F);
      xiB[t] = fmaf(kMom, xiB[t], (1.0f - kMom) * xaB[t]);
    }
  }

  // ---- scatter back: idx is a permutation, so rows go back where they came from ----
  const int rowA = ((g + n16 * sstep) % kM) * kCache + n16;
  const int rowB = ((g + (16 + n16) * sstep) % kM) * kCache + (16 + n16);
  float* pxA = x_out  + ((size_t)b * kN + rowA) * kE + n * kCE + eB;
  float* pxB = x_out  + ((size_t)b * kN + rowB) * kE + n * kCE + eB;
  float* paA = xa_out + ((size_t)b * kN + rowA) * kE + n * kCE + eB;
  float* paB = xa_out + ((size_t)b * kN + rowB) * kE + n * kCE + eB;
  #pragma unroll
  for (int c = 0; c < 4; ++c) {
    *reinterpret_cast<float4*>(pxA + c * 4) = make_float4(xiA[4*c+0], xiA[4*c+1], xiA[4*c+2], xiA[4*c+3]);
    *reinterpret_cast<float4*>(pxB + c * 4) = make_float4(xiB[4*c+0], xiB[4*c+1], xiB[4*c+2], xiB[4*c+3]);
    *reinterpret_cast<float4*>(paA + c * 4) = make_float4(xaA[4*c+0], xaA[4*c+1], xaA[4*c+2], xaA[4*c+3]);
    *reinterpret_cast<float4*>(paB + c * 4) = make_float4(xaB[4*c+0], xaB[4*c+1], xaB[4*c+2], xaB[4*c+3]);
  }
}

extern "C" void kernel_launch(void* const* d_in, const int* in_sizes, int n_in,
                              void* d_out, int out_size, void* d_ws, size_t ws_size,
                              hipStream_t stream) {
  (void)in_sizes; (void)n_in; (void)out_size; (void)ws_size;
  const float* x  = (const float*)d_in[0];   // [B,N,E] f32
  const float* xa = (const float*)d_in[1];   // [B,N,E] f32
  const float* W  = (const float*)d_in[2];   // [2, 2E]  f32

  const size_t tsz = (size_t)kB * kN * kE;   // 2,097,152 floats (8 MB)
  float* x1  = (float*)d_ws;                 // inter-layer ping-pong (16 MB of ws)
  float* xa1 = x1 + tsz;
  float* xo  = (float*)d_out;                // output: x then xa, concatenated flat
  float* xao = xo + tsz;

  dim3 grid(kB * kM * kNK), block(32);       // one wave32 per [32x32] tile
  // layer 0: stage 0 -> group stride s = 0
  hipLaunchKernelGGL(ncn_layer_kernel, grid, block, 0, stream, x,  xa,  W,          x1, xa1, 0);
  // layer 1: stage 1 -> s = (1 << 0) % m = 1
  hipLaunchKernelGGL(ncn_layer_kernel, grid, block, 0, stream, x1, xa1, W + 2 * kE, xo, xao, 1);
}
